// EdgePredictionHead_58815282151678
// MI455X (gfx1250) — matrix operand order repaired and checked by reading, not compiled, not run on hardware
//
#include <hip/hip_runtime.h>

typedef _Float16 h16;
typedef __attribute__((ext_vector_type(8)))  _Float16 v8h;
typedef __attribute__((ext_vector_type(16))) _Float16 v16h;
typedef __attribute__((ext_vector_type(8)))  float    f8;
typedef __attribute__((ext_vector_type(8)))  float    v8f;

#define N_NODES 1024
#define SDIM 256
#define VDIM 64
#define EDIM 128
#define NUM_MOL 32
#define ATOMS 32
#define EDGES 31744            // 32 * 32 * 31
#define EPM 992                // edges per molecule
#define TILES 1984             // EDGES / 16

// ---- workspace layout (bytes) ----
#define WS_WST   0                         // W_shared_t f16 [256][256]  131072 B
#define WS_WBT   131072                    // W_bond_t  f16 [256][128]    65536 B  (contig start)
#define WS_WB0T  (131072+65536)            // W_b0_t    f16 [256][256]   131072 B
#define WS_WB1T  (131072+65536+131072)     // W_b1_t    f16 [16][256]      8192 B  (contig end)
#define WS_ESYM  (WS_WB1T+8192)            // e_sym f16 [E][128]        8126464 B
#define WS_WD    (WS_ESYM+8126464)         // w_d  f32 [256]
#define WS_CRD   (WS_WD+1024)              // coords f32 [1024][3]
#define WS_SACT  (WS_CRD+12288)            // s_act f32 [1024][256]

// contiguous weight region copied to LDS by edge kernel
#define WCONTIG_BYTES (65536+131072+8192)  // 204800
#define FSCR_STRIDE 264                    // halves; 528B => 4-bank skew, conflict-free rows
#define EDGE_DYN_LDS (WCONTIG_BYTES + 2*16*FSCR_STRIDE*2)  // 221696

static __device__ __forceinline__ v8f wmma_f16(v16h a, v16h b, v8f c) {
  return __builtin_amdgcn_wmma_f32_16x16x32_f16(false, a, false, b, (short)0, c, false, false);
}
static __device__ __forceinline__ v16h cat8(v8h lo, v8h hi) {
  return __builtin_shufflevector(lo, hi, 0,1,2,3,4,5,6,7,8,9,10,11,12,13,14,15);
}
static __device__ __forceinline__ v16h load_a_frag(const h16* p) {
  // per ISA 7.12.2: elements 0..7 = K[kb..kb+7], 8..15 = K[kb+16..kb+23]
  v8h lo = *(const v8h*)p;
  v8h hi = *(const v8h*)(p + 16);
  return cat8(lo, hi);
}
static __device__ __forceinline__ float silu(float x) {
  return x * (1.0f / (1.0f + __expf(-x)));
}

// ---------------- prep: transpose + f32->f16 weights ----------------
__global__ void prep_weights(const float* __restrict__ Ws, const float* __restrict__ Wb,
                             const float* __restrict__ Wb0, const float* __restrict__ Wb1,
                             h16* __restrict__ wst, h16* __restrict__ wbt,
                             h16* __restrict__ wb0t, h16* __restrict__ wb1t,
                             float* __restrict__ wd) {
  int idx = blockIdx.x * 256 + threadIdx.x;               // 0..65535
  { int n = idx >> 8, k = idx & 255; wst[n*256 + k]  = (h16)Ws[k*256 + n];  }
  if (idx < 32768) { int n = idx >> 7, k = idx & 127; wbt[n*128 + k]  = (h16)Wb[k*256 + n];  }
  { int n = idx >> 8, k = idx & 255; wb0t[n*256 + k] = (h16)Wb0[k*256 + n]; }
  if (idx < 4096)  { int n = idx >> 8, k = idx & 255; wb1t[idx] = (n < 5) ? (h16)Wb1[k*5 + n] : (h16)0.0f; }
  if (idx < 256)   wd[idx] = Wb0[256*256 + idx];
}

// ---------------- e symmetrization (analytic reverse edge) ----------------
__global__ void esym_kernel(const float* __restrict__ e, const int* __restrict__ ei,
                            h16* __restrict__ esym) {
  int t = blockIdx.x * 256 + threadIdx.x;                 // E*128 threads exactly
  int eid = t >> 7;
  int k   = t & 127;
  int jn = ei[eid], in_ = ei[EDGES + eid];
  int jj = jn & 31, ii = in_ & 31, mol = jn >> 5;
  int rev = mol * EPM + ii * 31 + jj - (jj > ii ? 1 : 0);
  esym[t] = (h16)(0.5f * (e[t] + e[rev*EDIM + k]));
}

// ---------------- coords: p + v.Wc, per-molecule centering ----------------
__global__ void coords_kernel(const float* __restrict__ v, const float* __restrict__ p,
                              const float* __restrict__ wc, float* __restrict__ coords) {
  __shared__ float cbuf[96];
  __shared__ float mean[3];
  int mol = blockIdx.x, t = threadIdx.x;
  int a = t / 3, c = t - a * 3;
  int node = mol * ATOMS + a;
  if (t < 96) {
    float acc = p[node*3 + c];
    const float* vr = v + (node*3 + c) * VDIM;
    #pragma unroll 8
    for (int vd = 0; vd < VDIM; ++vd) acc += vr[vd] * wc[vd];
    cbuf[t] = acc;
  }
  __syncthreads();
  if (t < 3) {
    float s = 0.0f;
    for (int a2 = 0; a2 < ATOMS; ++a2) s += cbuf[a2*3 + t];
    mean[t] = s * (1.0f / (float)ATOMS);
  }
  __syncthreads();
  if (t < 96) coords[node*3 + c] = cbuf[t] - mean[c];
}

// ---------------- s_act = silu(s @ W_shared + b) via WMMA ----------------
__global__ __launch_bounds__(32) void sact_kernel(const float* __restrict__ s,
                                                  const h16* __restrict__ wst,
                                                  const float* __restrict__ bsh,
                                                  float* __restrict__ sact) {
  int lane = threadIdx.x, lane15 = lane & 15, hs = lane >> 4;
  int mtile = blockIdx.x;                                  // 64 tiles of 16 rows
  v8f zero = {0.f,0.f,0.f,0.f,0.f,0.f,0.f,0.f};
  v8f acc[16];
  #pragma unroll
  for (int i = 0; i < 16; ++i) acc[i] = zero;
  #pragma unroll
  for (int kt = 0; kt < 8; ++kt) {
    const float* ap = s + (mtile*16 + lane15) * SDIM + kt*32 + hs*8;
    f8 lo = *(const f8*)ap;
    f8 hi = *(const f8*)(ap + 16);
    v16h a = cat8(__builtin_convertvector(lo, v8h), __builtin_convertvector(hi, v8h));
    #pragma unroll
    for (int nt = 0; nt < 16; ++nt) {
      v16h b = load_a_frag(wst + (nt*16 + lane15) * SDIM + kt*32 + hs*8);
      acc[nt] = wmma_f16(a, b, acc[nt]);
    }
  }
  #pragma unroll
  for (int nt = 0; nt < 16; ++nt) {
    int col = nt*16 + lane15;
    float bb = bsh[col];
    #pragma unroll
    for (int r = 0; r < 8; ++r) {
      int row = mtile*16 + r + hs*8;
      sact[row*SDIM + col] = silu(acc[nt][r] + bb);
    }
  }
}

// ---------------- fused edge pipeline: 3 chained WMMA GEMMs ----------------
__global__ __launch_bounds__(64) void edge_kernel(
    const h16* __restrict__ wsrc,      // contiguous wbt|wb0t|wb1t (204800 B)
    const h16* __restrict__ esym,
    const float* __restrict__ sact,
    const float* __restrict__ coords,
    const int* __restrict__ ei,
    const float* __restrict__ b_bond,
    const float* __restrict__ b_b0,
    const float* __restrict__ wd,
    const float* __restrict__ b_b1,
    float* __restrict__ out) {
  extern __shared__ char smem[];
  h16* sWbt  = (h16*)smem;                                 // [256][128]
  h16* sWb0t = sWbt  + 256*128;                            // [256][256]
  h16* sWb1t = sWb0t + 256*256;                            // [16][256]
  h16* fscrA = sWb1t + 16*256;

  int tid = threadIdx.x, wave = tid >> 5, lane = tid & 31;
  int lane15 = lane & 15, hs = lane >> 4, rowoff = hs * 8;
  h16* fscr = fscrA + wave * (16 * FSCR_STRIDE);           // wave-private scratch

  // cooperative weight load into LDS (204800 B = 12800 uint4)
  {
    const uint4* src = (const uint4*)wsrc;
    uint4* dst = (uint4*)smem;
    for (int i = tid; i < 12800; i += 64) dst[i] = src[i];
  }
  __syncthreads();

  v8f zero = {0.f,0.f,0.f,0.f,0.f,0.f,0.f,0.f};
  int wlin = blockIdx.x * 2 + wave;                        // 248 waves total

  for (int it = 0; it < 8; ++it) {
    int tile = wlin * 8 + it;                              // 248*8 == TILES
    int base = tile * 16;

    // per-row edge metadata (lanes 0..15 carry rows 0..15; upper half mirrors)
    int eid = base + lane15;
    int jn = ei[eid], in_ = ei[EDGES + eid];
    float dx = coords[in_*3+0] - coords[jn*3+0];
    float dy = coords[in_*3+1] - coords[jn*3+1];
    float dz = coords[in_*3+2] - coords[jn*3+2];
    float dval = dx*dx + dy*dy + dz*dz;

    int ir[8], jr[8]; float dr[8];
    #pragma unroll
    for (int r = 0; r < 8; ++r) {
      ir[r] = __shfl(in_,  r + rowoff, 32);
      jr[r] = __shfl(jn,   r + rowoff, 32);
      dr[r] = __shfl(dval, r + rowoff, 32);
    }

    // ---- GEMM1: e_sym[16,128] @ W_bond -> acc1[16,256] ----
    v8f acc1[16];
    #pragma unroll
    for (int i = 0; i < 16; ++i) acc1[i] = zero;
    const h16* erow = esym + (size_t)(base + lane15) * EDIM;
    #pragma unroll
    for (int kt = 0; kt < 4; ++kt) {
      v16h a = load_a_frag(erow + kt*32 + hs*8);
      #pragma unroll
      for (int nt = 0; nt < 16; ++nt) {
        v16h b = load_a_frag(sWbt + (nt*16 + lane15) * EDIM + kt*32 + hs*8);
        acc1[nt] = wmma_f16(a, b, acc1[nt]);
      }
    }

    // ---- epilogue1: f = acc1 + s[i] + s[j] + b_bond -> f16 LDS scratch ----
    #pragma unroll
    for (int nt = 0; nt < 16; ++nt) {
      int col = nt*16 + lane15;
      float bb = b_bond[col];
      #pragma unroll
      for (int r = 0; r < 8; ++r) {
        float x = acc1[nt][r] + sact[ir[r]*SDIM + col] + sact[jr[r]*SDIM + col] + bb;
        fscr[(r + rowoff) * FSCR_STRIDE + col] = (h16)x;   // wave-private; DS in-order
      }
    }

    // ---- GEMM2: f[16,256] @ W_b0[0:256] -> acc2[16,256] ----
    v8f acc2[16];
    #pragma unroll
    for (int i = 0; i < 16; ++i) acc2[i] = zero;
    #pragma unroll
    for (int kt = 0; kt < 8; ++kt) {
      v16h a = load_a_frag(fscr + lane15 * FSCR_STRIDE + kt*32 + hs*8);
      #pragma unroll
      for (int nt = 0; nt < 16; ++nt) {
        v16h b = load_a_frag(sWb0t + (nt*16 + lane15) * SDIM + kt*32 + hs*8);
        acc2[nt] = wmma_f16(a, b, acc2[nt]);
      }
    }

    // ---- epilogue2: + d*w_d (rank-1, 257th K) + b_b0, silu -> h f16 ----
    #pragma unroll
    for (int nt = 0; nt < 16; ++nt) {
      int col = nt*16 + lane15;
      float bb = b_b0[col], wdc = wd[col];
      #pragma unroll
      for (int r = 0; r < 8; ++r) {
        float x = silu(acc2[nt][r] + bb + dr[r] * wdc);
        fscr[(r + rowoff) * FSCR_STRIDE + col] = (h16)x;
      }
    }

    // ---- GEMM3: h[16,256] @ W_b1(padded to 16 cols) ----
    v8f acc3 = zero;
    #pragma unroll
    for (int kt = 0; kt < 8; ++kt) {
      v16h a = load_a_frag(fscr + lane15 * FSCR_STRIDE + kt*32 + hs*8);
      v16h b = load_a_frag(sWb1t + lane15 * SDIM + kt*32 + hs*8);
      acc3 = wmma_f16(a, b, acc3);
    }

    if (lane15 < 5) {
      float bb = b_b1[lane15];
      #pragma unroll
      for (int r = 0; r < 8; ++r)
        out[(base + r + rowoff) * 5 + lane15] = acc3[r] + bb;
    }
  }
}

extern "C" void kernel_launch(void* const* d_in, const int* in_sizes, int n_in,
                              void* d_out, int out_size, void* d_ws, size_t ws_size,
                              hipStream_t stream) {
  (void)in_sizes; (void)n_in; (void)out_size; (void)ws_size;
  const float* s        = (const float*)d_in[0];
  const float* v        = (const float*)d_in[1];
  const float* p        = (const float*)d_in[2];
  const float* e        = (const float*)d_in[3];
  const int*   ei       = (const int*)  d_in[5];
  const float* W_shared = (const float*)d_in[6];
  const float* b_shared = (const float*)d_in[7];
  const float* W_coords = (const float*)d_in[8];
  const float* W_bond   = (const float*)d_in[9];
  const float* b_bond   = (const float*)d_in[10];
  const float* W_b0     = (const float*)d_in[11];
  const float* b_b0     = (const float*)d_in[12];
  const float* W_b1     = (const float*)d_in[13];
  const float* b_b1     = (const float*)d_in[14];
  float* out = (float*)d_out;

  char*  wsb    = (char*)d_ws;
  h16*   wst    = (h16*)  (wsb + WS_WST);
  h16*   wbt    = (h16*)  (wsb + WS_WBT);
  h16*   wb0t   = (h16*)  (wsb + WS_WB0T);
  h16*   wb1t   = (h16*)  (wsb + WS_WB1T);
  h16*   esym   = (h16*)  (wsb + WS_ESYM);
  float* wd     = (float*)(wsb + WS_WD);
  float* coords = (float*)(wsb + WS_CRD);
  float* sact   = (float*)(wsb + WS_SACT);

  prep_weights<<<256, 256, 0, stream>>>(W_shared, W_bond, W_b0, W_b1, wst, wbt, wb0t, wb1t, wd);
  esym_kernel<<<(EDGES * EDIM) / 256, 256, 0, stream>>>(e, ei, esym);
  coords_kernel<<<NUM_MOL, 96, 0, stream>>>(v, p, W_coords, coords);
  sact_kernel<<<N_NODES / 16, 32, 0, stream>>>(s, wst, b_shared, sact);

  (void)hipFuncSetAttribute((const void*)edge_kernel,
                            hipFuncAttributeMaxDynamicSharedMemorySize, EDGE_DYN_LDS);
  edge_kernel<<<124, 64, EDGE_DYN_LDS, stream>>>(wbt, esym, sact, coords, ei,
                                                 b_bond, b_b0, wd, b_b1, out);
}
